// QwenGroupedQueryAttention_19258633355374
// MI455X (gfx1250) — matrix operand order repaired and checked by reading, not compiled, not run on hardware
//
#include <hip/hip_runtime.h>
#include <hip/hip_bf16.h>
#include <math.h>

// ---------------------------------------------------------------------------
// Types for CDNA5 WMMA (wave32)
// ---------------------------------------------------------------------------
typedef __bf16 bf16_t;
typedef __attribute__((ext_vector_type(16))) __bf16 v16bf;
typedef __attribute__((ext_vector_type(8)))  float  v8f;

#define BATCH 2
#define TSEQ  2048
#define EMBED 2048
#define HQ    16
#define HKV   4
#define HD    128
#define NKV   (HKV * HD)     // 512
#define MROWS (BATCH * TSEQ) // 4096
#define NKT   (TSEQ / 32)    // 64 key tiles of 32

// ---------------------------------------------------------------------------
// Helpers
// ---------------------------------------------------------------------------
__device__ __forceinline__ bf16_t f2bf(float f) {
  union { float f; unsigned u; } v; v.f = f;
  unsigned r = (v.u + 0x7FFFu + ((v.u >> 16) & 1u)) >> 16;
  union { unsigned short s; bf16_t b; } o; o.s = (unsigned short)r;
  return o.b;
}

__device__ __forceinline__ v8f zero8() {
  v8f z = {0.f, 0.f, 0.f, 0.f, 0.f, 0.f, 0.f, 0.f};
  return z;
}

// A fragment (16x32, bf16, row-major source, lda in elements).
// Lane l holds row (l&15); elements 0..15 = K (l>>4)*16 + 0..15 (contig).
__device__ __forceinline__ v16bf load_a_frag(const bf16_t* __restrict__ A, int lda,
                                             int row0, int k0, int lane) {
  union { v16bf v; int4 q[2]; } u;
  const bf16_t* p = A + (size_t)(row0 + (lane & 15)) * lda + k0 + ((lane >> 4) << 4);
  u.q[0] = *reinterpret_cast<const int4*>(p);
  u.q[1] = *reinterpret_cast<const int4*>(p + 8);
  return u.v;
}

// B fragment (32x16, bf16) from an N-major matrix BT[N][K] (ldb in elements).
// Lane l holds column (l&15); elements 0..15 = K (l>>4)*16 + 0..15 (contig).
__device__ __forceinline__ v16bf load_b_frag(const bf16_t* __restrict__ BT, int ldb,
                                             int col0, int k0, int lane) {
  union { v16bf v; int4 q[2]; } u;
  const bf16_t* p = BT + (size_t)(col0 + (lane & 15)) * ldb + k0 + ((lane >> 4) << 4);
  u.q[0] = *reinterpret_cast<const int4*>(p);
  u.q[1] = *reinterpret_cast<const int4*>(p + 8);
  return u.v;
}

__device__ __forceinline__ v8f wmma_bf16(v16bf a, v16bf b, v8f c) {
  return __builtin_amdgcn_wmma_f32_16x16x32_bf16(false, a, false, b,
                                                 (short)0, c, false, false);
}

// CDNA5 async copy: 16 bytes per lane, global -> LDS, tracked by ASYNCcnt.
__device__ __forceinline__ void async_cp16(void* lds_ptr, const void* gptr) {
  unsigned lds_off = (unsigned)(size_t)lds_ptr; // low 32 bits of generic = LDS offset
  asm volatile("global_load_async_to_lds_b128 %0, %1, off"
               :: "v"(lds_off), "v"(gptr) : "memory");
}

// ---------------------------------------------------------------------------
// Precision conversion kernels
// ---------------------------------------------------------------------------
__global__ void k_f32_to_bf16(const float* __restrict__ in, bf16_t* __restrict__ out, int n) {
  int i = blockIdx.x * blockDim.x + threadIdx.x;
  if (i < n) out[i] = f2bf(in[i]);
}

// W[K][N] fp32 -> WT[N][K] bf16
__global__ void k_transpose_bf16(const float* __restrict__ W, bf16_t* __restrict__ WT,
                                 int K, int N) {
  int i = blockIdx.x * blockDim.x + threadIdx.x;
  if (i < K * N) {
    int k = i / N, n = i - k * N;
    WT[(size_t)n * K + k] = f2bf(W[i]);
  }
}

// ---------------------------------------------------------------------------
// WMMA GEMM: out[M,N] = A[M,K](bf16,row-major) * BT[N,K]^T (+bias)(+scale)
// MODE 0: Q (bias, *scale, bf16 row-major)   MODE 1: K (bias, bf16 row-major)
// MODE 2: V (bias, bf16 transposed Vt[b][g][d][t])   MODE 3: O-proj (fp32)
// Wave computes 16x64 tile; fragments register-double-buffered so loads for
// k+32 overlap the WMMAs for k.
// ---------------------------------------------------------------------------
template <int MODE>
__global__ __launch_bounds__(256) void k_gemm(const bf16_t* __restrict__ A,
                                              const bf16_t* __restrict__ BT,
                                              const float* __restrict__ bias,
                                              void* __restrict__ outp,
                                              int M, int N, int K, float scale) {
  const int lane = threadIdx.x & 31;
  const int wave = threadIdx.x >> 5;
  const int tilesN = N >> 6;
  const int total = (M >> 4) * tilesN;
  int t = blockIdx.x * (blockDim.x >> 5) + wave;
  if (t >= total) return;
  const int row0 = (t / tilesN) << 4;
  const int col0 = (t % tilesN) << 6;

  v8f acc[4];
  acc[0] = zero8(); acc[1] = zero8(); acc[2] = zero8(); acc[3] = zero8();

  v16bf a_cur = load_a_frag(A, K, row0, 0, lane);
  v16bf b_cur[4];
#pragma unroll
  for (int j = 0; j < 4; ++j) b_cur[j] = load_b_frag(BT, K, col0 + j * 16, 0, lane);

  for (int k0 = 0; k0 < K; k0 += 32) {
    const int kn = k0 + 32;
    v16bf a_nxt = a_cur;
    v16bf b_nxt[4] = {b_cur[0], b_cur[1], b_cur[2], b_cur[3]};
    if (kn < K) {
      __builtin_prefetch(A + (size_t)(row0 + (lane & 15)) * K + kn + 32, 0, 0);
      a_nxt = load_a_frag(A, K, row0, kn, lane);
#pragma unroll
      for (int j = 0; j < 4; ++j) b_nxt[j] = load_b_frag(BT, K, col0 + j * 16, kn, lane);
    }
#pragma unroll
    for (int j = 0; j < 4; ++j) acc[j] = wmma_bf16(a_cur, b_cur[j], acc[j]);
    a_cur = a_nxt;
#pragma unroll
    for (int j = 0; j < 4; ++j) b_cur[j] = b_nxt[j];
  }

  const int half = lane >> 4, cl = lane & 15;
#pragma unroll
  for (int j = 0; j < 4; ++j) {
    const int n = col0 + j * 16 + cl;
    float bv = 0.f;
    if (MODE != 3) bv = bias[n];
#pragma unroll
    for (int r = 0; r < 8; ++r) {
      const int m = row0 + half * 8 + r;
      float val = acc[j][r] + bv;
      if (MODE == 0) {
        ((bf16_t*)outp)[(size_t)m * N + n] = f2bf(val * scale);
      } else if (MODE == 1) {
        ((bf16_t*)outp)[(size_t)m * N + n] = f2bf(val);
      } else if (MODE == 2) {
        const int b = m >> 11, tt = m & 2047, g = n >> 7, d = n & 127;
        ((bf16_t*)outp)[(((size_t)(b * HKV + g) * HD + d) << 11) + tt] = f2bf(val);
      } else {
        ((float*)outp)[(size_t)m * N + n] = val;
      }
    }
  }
}

// ---------------------------------------------------------------------------
// Flash attention. One wave = one 16-query tile; 4 waves/block handle 4
// consecutive q-tiles of the SAME (batch, head) => identical K/V stream.
// K/V tiles (32 keys) are cooperatively staged into LDS with CDNA5 async
// global->LDS copies, double-buffered: next tile's copies are in flight
// while the current tile's 16 WMMAs execute.
// ---------------------------------------------------------------------------
#define WPB 4
__global__ __launch_bounds__(128) void k_attn(const bf16_t* __restrict__ Qb,
                                              const bf16_t* __restrict__ Kb,
                                              const bf16_t* __restrict__ Vt,
                                              bf16_t* __restrict__ Ob) {
  __shared__ bf16_t Ksh[2][32][128];  // [buf][key][d]   8KB per buffer
  __shared__ bf16_t Vsh[2][128][32];  // [buf][d][t]     8KB per buffer
  __shared__ float  Ssh[WPB][16][33];
  __shared__ bf16_t Psh[WPB][16][40]; // row stride 80B (16B aligned)
  __shared__ float  csh[WPB][16];
  __shared__ float  lsh[WPB][16];

  const int tid  = threadIdx.x;
  const int lane = tid & 31;
  const int w    = tid >> 5;
  const int QT   = TSEQ / 16;
  const int gw   = blockIdx.x * WPB + w;
  const int qt = gw % QT;
  const int h  = (gw / QT) % HQ;     // uniform per block (QT % WPB == 0)
  const int b  = gw / (QT * HQ);
  const int g  = h >> 2;
  const int half = lane >> 4, cl = lane & 15;

  const bf16_t* Qrow  = Qb + (size_t)(b * TSEQ + qt * 16) * EMBED + h * HD;
  const bf16_t* Kbase = Kb + (size_t)(b * TSEQ) * NKV + g * HD;
  const bf16_t* Vbase = Vt + (size_t)(b * HKV + g) * HD * TSEQ;

  // Q tile (16 x 128) as 4 A-fragments, kept in registers.
  v16bf qf[4];
#pragma unroll
  for (int i = 0; i < 4; ++i) qf[i] = load_a_frag(Qrow, EMBED, 0, i * 32, lane);

  v8f oacc[8];
#pragma unroll
  for (int i = 0; i < 8; ++i) oacc[i] = zero8();

  float m_old = -3.0e38f, l_run = 0.f; // lanes 0..15 own query row = lane

  // Stage one 32-key K/V tile into LDS buffer `buf` (8 async ops per wave).
  auto stage = [&](int kt, int buf) {
#pragma unroll
    for (int i = 0; i < 4; ++i) {       // K tile: 512 x 16B chunks over 128 thr
      const int c = tid + i * 128;
      const int r = c >> 4, cc = c & 15;
      async_cp16(&Ksh[buf][r][cc * 8],
                 Kbase + (size_t)(kt * 32 + r) * NKV + cc * 8);
    }
#pragma unroll
    for (int i = 0; i < 4; ++i) {       // V tile: [d][32t]
      const int c = tid + i * 128;
      const int d = c >> 2, tc = c & 3;
      async_cp16(&Vsh[buf][d][tc * 8],
                 Vbase + (size_t)d * TSEQ + kt * 32 + tc * 8);
    }
  };

  stage(0, 0);

  for (int kt = 0; kt < NKT; ++kt) {
    const int buf = kt & 1;
    if (kt + 1 < NKT) {
      stage(kt + 1, buf ^ 1);
      asm volatile("s_wait_asynccnt 8" ::: "memory");  // this tile's 8 done
    } else {
      asm volatile("s_wait_asynccnt 0" ::: "memory");
    }
    __syncthreads();  // all threads' staging for `buf` complete

    // ---- S = Q * K^T for 32 keys (2 x 16-column WMMA tiles) ----
#pragma unroll
    for (int ct = 0; ct < 2; ++ct) {
      v8f s = zero8();
#pragma unroll
      for (int kk = 0; kk < 4; ++kk) {
        union { v16bf v; int4 q[2]; } u;
        const bf16_t* p = &Ksh[buf][ct * 16 + cl][kk * 32 + half * 16];
        u.q[0] = *reinterpret_cast<const int4*>(p);
        u.q[1] = *reinterpret_cast<const int4*>(p + 8);
        s = wmma_bf16(qf[kk], u.v, s);
      }
#pragma unroll
      for (int r = 0; r < 8; ++r)
        Ssh[w][half * 8 + r][ct * 16 + cl] = s[r];
    }
    __syncthreads();

    // ---- online softmax: lanes 0..15, one query row each ----
    if (lane < 16) {
      const int row = lane;
      float sv[32];
      float rmax = -3.0e38f;
#pragma unroll
      for (int j = 0; j < 32; ++j) { sv[j] = Ssh[w][row][j]; rmax = fmaxf(rmax, sv[j]); }
      const float m_new = fmaxf(m_old, rmax);
      const float c = __expf(m_old - m_new);
      float rs = 0.f;
#pragma unroll
      for (int j = 0; j < 32; ++j) {
        const float p = __expf(sv[j] - m_new);
        rs += p;
        Psh[w][row][j] = f2bf(p);
      }
      l_run = l_run * c + rs;
      m_old = m_new;
      csh[w][row] = c;
    }
    __syncthreads();

    // ---- rescale running O, then O += P * V ----
    float cr[8];
#pragma unroll
    for (int r = 0; r < 8; ++r) cr[r] = csh[w][half * 8 + r];
#pragma unroll
    for (int nt = 0; nt < 8; ++nt)
#pragma unroll
      for (int r = 0; r < 8; ++r) oacc[nt][r] *= cr[r];

    union { v16bf v; int4 q[2]; } pu;  // P as A-fragment straight from LDS
    const bf16_t* prow = &Psh[w][cl][0];
    pu.q[0] = *reinterpret_cast<const int4*>(prow + half * 16);
    pu.q[1] = *reinterpret_cast<const int4*>(prow + half * 16 + 8);

#pragma unroll
    for (int nt = 0; nt < 8; ++nt) {
      union { v16bf v; int4 q[2]; } vu;
      const bf16_t* p = &Vsh[buf][nt * 16 + cl][half * 16];
      vu.q[0] = *reinterpret_cast<const int4*>(p);
      vu.q[1] = *reinterpret_cast<const int4*>(p + 8);
      oacc[nt] = wmma_bf16(pu.v, vu.v, oacc[nt]);
    }
    __syncthreads();  // protects Psh and the buffer re-issued next iteration
  }

  if (lane < 16) lsh[w][lane] = 1.0f / l_run;
  __syncthreads();

  float li[8];
#pragma unroll
  for (int r = 0; r < 8; ++r) li[r] = lsh[w][half * 8 + r];

  bf16_t* Orow = Ob + (size_t)(b * TSEQ + qt * 16) * EMBED + h * HD;
#pragma unroll
  for (int nt = 0; nt < 8; ++nt)
#pragma unroll
    for (int r = 0; r < 8; ++r)
      Orow[(size_t)(half * 8 + r) * EMBED + nt * 16 + cl] = f2bf(oacc[nt][r] * li[r]);
}

// ---------------------------------------------------------------------------
// Launcher
// ---------------------------------------------------------------------------
extern "C" void kernel_launch(void* const* d_in, const int* in_sizes, int n_in,
                              void* d_out, int out_size, void* d_ws, size_t ws_size,
                              hipStream_t stream) {
  (void)in_sizes; (void)n_in; (void)out_size; (void)ws_size;
  const float* x  = (const float*)d_in[0];
  const float* Wq = (const float*)d_in[1];
  const float* bq = (const float*)d_in[2];
  const float* Wk = (const float*)d_in[3];
  const float* bk = (const float*)d_in[4];
  const float* Wv = (const float*)d_in[5];
  const float* bv = (const float*)d_in[6];
  const float* Wo = (const float*)d_in[7];
  float* out = (float*)d_out;

  char* ws = (char*)d_ws;
  size_t off = 0;
  auto carve = [&](size_t bytes) -> char* {
    char* p = ws + off;
    off += (bytes + 255) & ~(size_t)255;
    return p;
  };

  bf16_t* x_bf   = (bf16_t*)carve((size_t)MROWS * EMBED * 2);
  bf16_t* WqT    = (bf16_t*)carve((size_t)EMBED * EMBED * 2);
  bf16_t* WkT    = (bf16_t*)carve((size_t)NKV * EMBED * 2);
  bf16_t* WvT    = (bf16_t*)carve((size_t)NKV * EMBED * 2);
  bf16_t* WoT    = (bf16_t*)carve((size_t)EMBED * EMBED * 2);
  bf16_t* Q_bf   = (bf16_t*)carve((size_t)MROWS * EMBED * 2);
  bf16_t* K_bf   = (bf16_t*)carve((size_t)MROWS * NKV * 2);
  bf16_t* Vt_bf  = (bf16_t*)carve((size_t)MROWS * NKV * 2);
  bf16_t* att_bf = (bf16_t*)carve((size_t)MROWS * EMBED * 2);

  // 1) precision conversion + weight transposition
  {
    int n = MROWS * EMBED;
    k_f32_to_bf16<<<(n + 255) / 256, 256, 0, stream>>>(x, x_bf, n);
  }
  k_transpose_bf16<<<(EMBED * EMBED + 255) / 256, 256, 0, stream>>>(Wq, WqT, EMBED, EMBED);
  k_transpose_bf16<<<(EMBED * NKV   + 255) / 256, 256, 0, stream>>>(Wk, WkT, EMBED, NKV);
  k_transpose_bf16<<<(EMBED * NKV   + 255) / 256, 256, 0, stream>>>(Wv, WvT, EMBED, NKV);
  k_transpose_bf16<<<(EMBED * EMBED + 255) / 256, 256, 0, stream>>>(Wo, WoT, EMBED, EMBED);

  const float scale = 1.0f / sqrtf((float)HD);

  // 2) projections
  {
    int tiles = (MROWS / 16) * (EMBED / 64);
    k_gemm<0><<<tiles / 8, 256, 0, stream>>>(x_bf, WqT, bq, (void*)Q_bf,
                                             MROWS, EMBED, EMBED, scale);
  }
  {
    int tiles = (MROWS / 16) * (NKV / 64);
    k_gemm<1><<<tiles / 8, 256, 0, stream>>>(x_bf, WkT, bk, (void*)K_bf,
                                             MROWS, NKV, EMBED, 1.f);
    k_gemm<2><<<tiles / 8, 256, 0, stream>>>(x_bf, WvT, bv, (void*)Vt_bf,
                                             MROWS, NKV, EMBED, 1.f);
  }

  // 3) attention core (async LDS staging + WMMA)
  {
    int waves = BATCH * HQ * (TSEQ / 16); // 4096
    k_attn<<<waves / WPB, WPB * 32, 0, stream>>>(Q_bf, K_bf, Vt_bf, att_bf);
  }

  // 4) output projection -> fp32 result
  {
    int tiles = (MROWS / 16) * (EMBED / 64);
    k_gemm<3><<<tiles / 8, 256, 0, stream>>>(att_bf, WoT, nullptr, (void*)out,
                                             MROWS, EMBED, EMBED, 1.f);
  }
}
